// SS2D_40450001994051
// MI455X (gfx1250) — matrix operand order repaired and checked by reading, not compiled, not run on hardware
//
#include <hip/hip_runtime.h>
#include <math.h>

// ---------------- problem constants ----------------
#define B_  8
#define H_  56
#define W_  56
#define L_  (H_ * W_)          // 3136
#define DM_ 96
#define DS_ 16
#define DI_ 192
#define R_  6
#define K_  4
#define XC_ 48                 // padded x_dbl row (38 used)
#define M_  (B_ * L_)          // 25088 rows for in/out proj GEMMs

typedef __attribute__((ext_vector_type(16))) _Float16 v16h;
typedef __attribute__((ext_vector_type(8)))  _Float16 v8h;
typedef __attribute__((ext_vector_type(8)))  float    v8f;

// ---------------- direction mapping (cross-scan) ----------------
__device__ __forceinline__ int map_dir(int k, int l) {
    switch (k) {
        case 0: return l;
        case 1: { int h = l % H_; int w = l / H_; return h * W_ + w; }
        case 2: return L_ - 1 - l;
        default: { int lr = L_ - 1 - l; int h = lr % H_; int w = lr / H_; return h * W_ + w; }
    }
}

// ---------------- WMMA fragment loaders (f16 operands, pre-converted) ----------------
// A (16x32 f16): lane 0-15 -> row M=lane&15, K runs [0..7],[16..23]; lanes 16-31 same rows,
// K runs [8..15],[24..31]. rowptr = &A[row][0], K-contiguous, 16B-aligned runs.
__device__ __forceinline__ v16h load_a_h(const _Float16* __restrict__ rowptr, int kk, int lane) {
    const int kb = (lane < 16) ? 0 : 8;
    const v8h lo = *(const v8h*)(rowptr + kk + kb);
    const v8h hi = *(const v8h*)(rowptr + kk + kb + 16);
    return __builtin_shufflevector(lo, hi, 0,1,2,3,4,5,6,7,8,9,10,11,12,13,14,15);
}
// B (32x16 f16): lane holds column N=lane&15, K runs [0..7],[16..23] (lanes 0-15) or [8..15],[24..31] (lanes 16-31).
__device__ __forceinline__ v16h load_b_h(const _Float16* __restrict__ colptr, int kk, int lane) {
    const int kb = (lane < 16) ? 0 : 8;
    const v8h lo = *(const v8h*)(colptr + kk + kb);
    const v8h hi = *(const v8h*)(colptr + kk + kb + 16);
    return __builtin_shufflevector(lo, hi, 0,1,2,3,4,5,6,7,8,9,10,11,12,13,14,15);
}
// C/D (16x16 f32, 8 VGPRs): lane -> n = n0 + (lane&15), rows m = m0 + r + 8*(lane>>4).
__device__ __forceinline__ void store_tile(float* __restrict__ out, int ldo, int m0, int n0,
                                           int lane, const v8f& c) {
    const int n = n0 + (lane & 15);
    const int mbase = m0 + ((lane >> 4) << 3);
#pragma unroll
    for (int r = 0; r < 8; ++r) out[(size_t)(mbase + r) * ldo + n] = c[r];
}

// ---------------- prep: f32 -> f16 converters ----------------
__global__ void k_f32_to_f16(const float* __restrict__ in, _Float16* __restrict__ out, int n) {
    const int i = blockIdx.x * blockDim.x + threadIdx.x;
    if (i < n) out[i] = (_Float16)in[i];
}
// x_proj_weight (K,38,DI) -> padded (K,48,DI) f16 with zero rows c>=38
__global__ void k_cvt_xpw(const float* __restrict__ in, _Float16* __restrict__ out) {
    const int i = blockIdx.x * blockDim.x + threadIdx.x;
    if (i >= K_ * XC_ * DI_) return;
    const int dd = i % DI_;
    const int c  = (i / DI_) % XC_;
    const int k  = i / (DI_ * XC_);
    out[i] = (c < R_ + 2 * DS_) ? (_Float16)in[((size_t)k * (R_ + 2 * DS_) + c) * DI_ + dd]
                                : (_Float16)0.0f;
}

// ---------------- kernel 1: in_proj GEMM (M_ x 96)f16 @ (192 x 96)^T f16 -> yin f32 ------
__global__ void k_inproj(const _Float16* __restrict__ x, const _Float16* __restrict__ w,
                         float* __restrict__ yin) {
    const int wid  = (blockIdx.x * blockDim.x + threadIdx.x) >> 5;
    const int lane = threadIdx.x & 31;
    const int nt = DI_ / 16, mt = M_ / 16;
    if (wid >= mt * nt) return;
    const int n0 = (wid % nt) * 16;
    const int m0 = (wid / nt) * 16;
    const _Float16* arow = x + (size_t)(m0 + (lane & 15)) * DM_;
    const _Float16* bcol = w + (size_t)(n0 + (lane & 15)) * DM_;
    v8f c = {};
#pragma unroll
    for (int kk = 0; kk < DM_; kk += 32) {
        v16h a = load_a_h(arow, kk, lane);
        v16h b = load_b_h(bcol, kk, lane);
        c = __builtin_amdgcn_wmma_f32_16x16x32_f16(false, a, false, b, (short)0, c, false, false);
    }
    store_tile(yin, DI_, m0, n0, lane, c);
}

// ---------------- kernel 2: depthwise 3x3 conv + bias + SiLU (writes f32 + f16 mirror) ----
__global__ void k_conv_silu(const float* __restrict__ yin, const float* __restrict__ cw,
                            const float* __restrict__ cb, float* __restrict__ xconv,
                            _Float16* __restrict__ xconv_h) {
    const size_t idx = (size_t)blockIdx.x * blockDim.x + threadIdx.x;
    if (idx >= (size_t)B_ * L_ * DI_) return;
    const int d = idx % DI_;
    const int w = (idx / DI_) % W_;
    const int h = (idx / ((size_t)DI_ * W_)) % H_;
    const int b = idx / ((size_t)DI_ * L_);
    float acc = cb[d];
#pragma unroll
    for (int kh = 0; kh < 3; ++kh) {
        const int hh = h + kh - 1;
        if (hh < 0 || hh >= H_) continue;
#pragma unroll
        for (int kw = 0; kw < 3; ++kw) {
            const int ww = w + kw - 1;
            if (ww < 0 || ww >= W_) continue;
            acc += cw[d * 9 + kh * 3 + kw] * yin[(((size_t)b * H_ + hh) * W_ + ww) * DI_ + d];
        }
    }
    const float s = acc / (1.0f + __expf(-acc));   // SiLU
    xconv[idx] = s;
    xconv_h[idx] = (_Float16)s;
}

// ---------------- kernel 3: x_dbl GEMM (gathered rows, padded f16 weights) ----------------
__global__ void k_xdbl(const _Float16* __restrict__ xconv_h, const _Float16* __restrict__ xpw_h,
                       float* __restrict__ xdbl) {
    const int wid  = (blockIdx.x * blockDim.x + threadIdx.x) >> 5;
    const int lane = threadIdx.x & 31;
    const int nt = XC_ / 16, mt = L_ / 16;
    if (wid >= B_ * K_ * mt * nt) return;
    const int n0 = (wid % nt) * 16;
    const int m0 = ((wid / nt) % mt) * 16;
    const int bk = wid / (nt * mt);
    const int b = bk >> 2, k = bk & 3;
    const int pos = map_dir(k, m0 + (lane & 15));
    const _Float16* arow = xconv_h + ((size_t)b * L_ + pos) * DI_;
    const _Float16* bcol = xpw_h + ((size_t)k * XC_ + (n0 + (lane & 15))) * DI_;
    v8f c = {};
#pragma unroll
    for (int kk = 0; kk < DI_; kk += 32) {
        v16h a = load_a_h(arow, kk, lane);
        v16h bb = load_b_h(bcol, kk, lane);
        c = __builtin_amdgcn_wmma_f32_16x16x32_f16(false, a, false, bb, (short)0, c, false, false);
    }
    float* out = xdbl + (size_t)(b * K_ + k) * L_ * XC_;
    store_tile(out, XC_, m0, n0, lane, c);
}

// ---------------- kernel 4: selective scan (async row staging into LDS) ----------------
__global__ void __launch_bounds__(DI_) k_scan(const float* __restrict__ xconv,
                                              const float* __restrict__ xdbl,
                                              const float* __restrict__ dtw,
                                              const float* __restrict__ dtb,
                                              const float* __restrict__ A_logs,
                                              const float* __restrict__ Ds,
                                              float* __restrict__ ysbuf) {
    const int bk = blockIdx.x;
    const int b = bk >> 2, k = bk & 3;
    const int d = threadIdx.x;                            // 0..191
    __shared__ float row[XC_];

    float a[DS_], h[DS_], w6[R_];
#pragma unroll
    for (int n = 0; n < DS_; ++n) {
        a[n] = -__expf(A_logs[((size_t)(k * DI_ + d)) * DS_ + n]);
        h[n] = 0.0f;
    }
#pragma unroll
    for (int r = 0; r < R_; ++r) w6[r] = dtw[((size_t)(k * DI_ + d)) * R_ + r];
    const float bias = dtb[k * DI_ + d];
    const float Dd = Ds[k * DI_ + d];

    const float* xrow = xdbl + (size_t)(b * K_ + k) * L_ * XC_;
    for (int l = 0; l < L_; ++l) {
        __syncthreads();                                   // previous row fully consumed
        if (d < XC_ / 4) {
            // async copy 16B/lane: global (dt,B,C) row -> LDS, tracked by ASYNCcnt
            const unsigned lds_dst = (unsigned)(size_t)(&row[d * 4]);
            const float* gsrc = xrow + (size_t)l * XC_ + d * 4;
            asm volatile("global_load_async_to_lds_b128 %0, %1, off"
                         :: "v"(lds_dst), "v"(gsrc) : "memory");
        }
        if (d == 0 && l + 8 < L_)
            __builtin_prefetch(xrow + (size_t)(l + 8) * XC_, 0, 0);   // global_prefetch_b8
        asm volatile("s_wait_asynccnt 0x0" ::: "memory");
        __syncthreads();

        const int pos = map_dir(k, l);
        const float u = xconv[((size_t)b * L_ + pos) * DI_ + d];
        float xr = bias;
#pragma unroll
        for (int r = 0; r < R_; ++r) xr += row[r] * w6[r];
        const float delta = (xr > 20.0f) ? xr : log1pf(__expf(xr));   // softplus
        const float du = delta * u;
        float y = 0.0f;
#pragma unroll
        for (int n = 0; n < DS_; ++n) {
            h[n] = __expf(delta * a[n]) * h[n] + du * row[R_ + n];
            y += h[n] * row[R_ + DS_ + n];
        }
        ysbuf[(((size_t)k * B_ + b) * L_ + pos) * DI_ + d] = y + u * Dd;
    }
}

// ---------------- kernel 5: 4-way merge + LayerNorm + GELU -> f16 ----------------
__global__ void k_merge_ln_gelu(const float* __restrict__ ysbuf, const float* __restrict__ g,
                                const float* __restrict__ bb, _Float16* __restrict__ ynorm) {
    const int rowi = blockIdx.x;                           // b*L + pos
    const int tid = threadIdx.x;                           // 0..255
    __shared__ float ssum[256], ssq[256];
    const size_t kstride = (size_t)B_ * L_ * DI_;
    float v = 0.0f;
    if (tid < DI_) {
        const size_t base = (size_t)rowi * DI_ + tid;
#pragma unroll
        for (int k = 0; k < K_; ++k) v += ysbuf[(size_t)k * kstride + base];
    }
    ssum[tid] = v;
    ssq[tid]  = v * v;
    __syncthreads();
    for (int s = 128; s > 0; s >>= 1) {
        if (tid < s) { ssum[tid] += ssum[tid + s]; ssq[tid] += ssq[tid + s]; }
        __syncthreads();
    }
    const float mean = ssum[0] * (1.0f / DI_);
    const float var  = ssq[0] * (1.0f / DI_) - mean * mean;
    if (tid < DI_) {
        const float rstd = rsqrtf(var + 1e-5f);
        const float yn = (v - mean) * rstd * g[tid] + bb[tid];
        const float ge = 0.5f * yn * (1.0f + erff(yn * 0.70710678118654752f));
        ynorm[(size_t)rowi * DI_ + tid] = (_Float16)ge;
    }
}

// ---------------- kernel 6: out_proj GEMM (M_ x 192)f16 @ (96 x 192)^T f16 -> d_out -------
__global__ void k_outproj(const _Float16* __restrict__ ynorm, const _Float16* __restrict__ w,
                          float* __restrict__ out) {
    const int wid  = (blockIdx.x * blockDim.x + threadIdx.x) >> 5;
    const int lane = threadIdx.x & 31;
    const int nt = DM_ / 16, mt = M_ / 16;
    if (wid >= mt * nt) return;
    const int n0 = (wid % nt) * 16;
    const int m0 = (wid / nt) * 16;
    const _Float16* arow = ynorm + (size_t)(m0 + (lane & 15)) * DI_;
    const _Float16* bcol = w + (size_t)(n0 + (lane & 15)) * DI_;
    v8f c = {};
#pragma unroll
    for (int kk = 0; kk < DI_; kk += 32) {
        v16h a = load_a_h(arow, kk, lane);
        v16h b = load_b_h(bcol, kk, lane);
        c = __builtin_amdgcn_wmma_f32_16x16x32_f16(false, a, false, b, (short)0, c, false, false);
    }
    store_tile(out, DM_, m0, n0, lane, c);
}

// ---------------- host launch ----------------
extern "C" void kernel_launch(void* const* d_in, const int* in_sizes, int n_in,
                              void* d_out, int out_size, void* d_ws, size_t ws_size,
                              hipStream_t stream) {
    const float* x    = (const float*)d_in[0];   // (B,H,W,96)
    const float* ipw  = (const float*)d_in[1];   // (192,96)
    const float* cw   = (const float*)d_in[2];   // (192,1,3,3)
    const float* cb   = (const float*)d_in[3];   // (192,)
    const float* xpw  = (const float*)d_in[4];   // (4,38,192)
    const float* dtw  = (const float*)d_in[5];   // (4,192,6)
    const float* dtb  = (const float*)d_in[6];   // (4,192)
    const float* alog = (const float*)d_in[7];   // (768,16)
    const float* Dsp  = (const float*)d_in[8];   // (768,)
    const float* lng  = (const float*)d_in[9];   // (192,)
    const float* lnb  = (const float*)d_in[10];  // (192,)
    const float* opw  = (const float*)d_in[11];  // (96,192)
    float* out = (float*)d_out;

    // workspace: yin | xconv | xdbl | ysbuf(4x) | ynorm(f16) | xh | xconv_h | weights(f16)
    const size_t szF = (size_t)B_ * L_ * DI_ * sizeof(float);     // 19,267,584 B
    const size_t szH = (size_t)B_ * L_ * DI_ * sizeof(_Float16);  //  9,633,792 B
    const size_t szXh = (size_t)M_ * DM_ * sizeof(_Float16);      //  4,816,896 B
    char* ws = (char*)d_ws;
    float*    yin     = (float*)ws;
    float*    xconv   = (float*)(ws + szF);
    float*    xdbl    = (float*)(ws + 2 * szF);                   // B*K*L*48 f32 == szF
    float*    ysb     = (float*)(ws + 3 * szF);                   // 4*szF
    _Float16* ynorm   = (_Float16*)(ws + 7 * szF);
    _Float16* xh      = (_Float16*)(ws + 7 * szF + szH);
    _Float16* xconv_h = (_Float16*)(ws + 7 * szF + szH + szXh);
    _Float16* ipw_h   = (_Float16*)(ws + 7 * szF + 2 * szH + szXh);
    _Float16* xpw_h   = ipw_h + DI_ * DM_;                        // padded (4,48,192)
    _Float16* opw_h   = xpw_h + K_ * XC_ * DI_;

    // 0) one-shot f16 conversions (weights + input activations)
    hipLaunchKernelGGL(k_f32_to_f16, dim3((M_ * DM_ + 255) / 256), dim3(256), 0, stream,
                       x, xh, M_ * DM_);
    hipLaunchKernelGGL(k_f32_to_f16, dim3((DI_ * DM_ + 255) / 256), dim3(256), 0, stream,
                       ipw, ipw_h, DI_ * DM_);
    hipLaunchKernelGGL(k_f32_to_f16, dim3((DM_ * DI_ + 255) / 256), dim3(256), 0, stream,
                       opw, opw_h, DM_ * DI_);
    hipLaunchKernelGGL(k_cvt_xpw, dim3((K_ * XC_ * DI_ + 255) / 256), dim3(256), 0, stream,
                       xpw, xpw_h);
    // 1) in_proj GEMM: 1568*12 = 18816 waves -> 2352 blocks x 256 (exact, EXEC all-1)
    hipLaunchKernelGGL(k_inproj, dim3(2352), dim3(256), 0, stream, xh, ipw_h, yin);
    // 2) depthwise conv + SiLU (f32 + f16 mirror)
    hipLaunchKernelGGL(k_conv_silu, dim3((B_ * L_ * DI_) / 256), dim3(256), 0, stream,
                       yin, cw, cb, xconv, xconv_h);
    // 3) x_dbl GEMM: 8*4*196*3 = 18816 waves -> 2352 blocks
    hipLaunchKernelGGL(k_xdbl, dim3(2352), dim3(256), 0, stream, xconv_h, xpw_h, xdbl);
    // 4) selective scan: one block per (b,k)
    hipLaunchKernelGGL(k_scan, dim3(B_ * K_), dim3(DI_), 0, stream,
                       xconv, xdbl, dtw, dtb, alog, Dsp, ysb);
    // 5) merge + LayerNorm + GELU
    hipLaunchKernelGGL(k_merge_ln_gelu, dim3(B_ * L_), dim3(256), 0, stream,
                       ysb, lng, lnb, ynorm);
    // 6) out_proj GEMM: 1568*6 = 9408 waves -> 1176 blocks
    hipLaunchKernelGGL(k_outproj, dim3(1176), dim3(256), 0, stream, ynorm, opw_h, out);
}
